// BitMambaBlock_30030411334045
// MI455X (gfx1250) — 8-bit matrix operand order repaired and checked by reading, compiled, not run on hardware
//
#include <hip/hip_runtime.h>
#include <hip/hip_bf16.h>
#include <cstdint>

#define DIM      1024
#define D_STATE  16
#define D_CONV   4
#define D_INNER  2048
#define DT_RANK  64
#define BATCH    2
#define SEQ      1024
#define ROWS     (BATCH * SEQ)
#define XDBL_W   (DT_RANK + 2 * D_STATE)

typedef __attribute__((ext_vector_type(8))) int v8i;

__device__ __forceinline__ float block_reduce_sum(float v, float* sm) {
    int t = threadIdx.x;
    sm[t] = v; __syncthreads();
    for (int s = 128; s > 0; s >>= 1) {
        if (t < s) sm[t] += sm[t + s];
        __syncthreads();
    }
    float r = sm[0]; __syncthreads();
    return r;
}

__device__ __forceinline__ float block_reduce_max(float v, float* sm) {
    int t = threadIdx.x;
    sm[t] = v; __syncthreads();
    for (int s = 128; s > 0; s >>= 1) {
        if (t < s) sm[t] = fmaxf(sm[t], sm[t + s]);
        __syncthreads();
    }
    float r = sm[0]; __syncthreads();
    return r;
}

__global__ void absmean_partial(const float* __restrict__ w, long n,
                                float* __restrict__ partial) {
    __shared__ float sm[256];
    float s = 0.f;
    for (long i = (long)blockIdx.x * 256 + threadIdx.x; i < n; i += 256L * gridDim.x)
        s += fabsf(w[i]);
    s = block_reduce_sum(s, sm);
    if (threadIdx.x == 0) partial[blockIdx.x] = s;
}

__global__ void finalize_scales(const float* __restrict__ partial,
                                float* __restrict__ wscale) {
    __shared__ float sm[256];
    const long counts[3] = { (long)(2 * D_INNER) * DIM,
                             (long)XDBL_W * D_INNER,
                             (long)DIM * D_INNER };
    for (int seg = 0; seg < 3; ++seg) {
        float v = partial[seg * 256 + threadIdx.x];
        v = block_reduce_sum(v, sm);
        if (threadIdx.x == 0)
            wscale[seg] = fmaxf(v / (float)counts[seg], 1e-5f);
        __syncthreads();
    }
}

__global__ void quant_weight(const float* __restrict__ w, long n,
                             const float* __restrict__ wscale,
                             int8_t* __restrict__ q) {
    const float inv = 1.0f / *wscale;
    for (long i = (long)blockIdx.x * blockDim.x + threadIdx.x; i < n;
         i += (long)blockDim.x * gridDim.x) {
        float r = rintf(w[i] * inv);
        r = fminf(1.f, fmaxf(-1.f, r));
        q[i] = (int8_t)r;
    }
}

__global__ void ln_act_quant(const float* __restrict__ X, int W,
                             const float* __restrict__ rms_w, int do_rms,
                             int8_t* __restrict__ Q,
                             float* __restrict__ inv_scale) {
    __shared__ float sm[256];
    const int t = threadIdx.x;
    const long row = blockIdx.x;
    const float* x = X + row * (long)W;
    const int vper = W >> 10;
    float v[8];

    float ss = 0.f;
    for (int i = 0; i < vper; ++i) {
        float4 f = ((const float4*)x)[i * 256 + t];
        v[4*i+0] = f.x; v[4*i+1] = f.y; v[4*i+2] = f.z; v[4*i+3] = f.w;
        ss += f.x*f.x + f.y*f.y + f.z*f.z + f.w*f.w;
    }

    if (do_rms) {
        ss = block_reduce_sum(ss, sm);
        const float r = rsqrtf(ss / (float)W + 1e-6f);
        for (int i = 0; i < vper; ++i) {
            float4 rw = ((const float4*)rms_w)[i * 256 + t];
            v[4*i+0] *= rw.x * r; v[4*i+1] *= rw.y * r;
            v[4*i+2] *= rw.z * r; v[4*i+3] *= rw.w * r;
        }
    }

    float s = 0.f;
    for (int i = 0; i < 4 * vper; ++i) s += v[i];
    s = block_reduce_sum(s, sm);
    const float mu = s / (float)W;

    float vs = 0.f;
    for (int i = 0; i < 4 * vper; ++i) { float d = v[i] - mu; vs += d * d; }
    vs = block_reduce_sum(vs, sm);
    const float rstd = rsqrtf(vs / (float)W + 1e-5f);

    float mx = 0.f;
    for (int i = 0; i < 4 * vper; ++i) {
        v[i] = (v[i] - mu) * rstd;
        mx = fmaxf(mx, fabsf(v[i]));
    }
    mx = block_reduce_max(mx, sm);
    mx = fmaxf(mx, 1e-5f);
    const float scale = 127.0f / mx;

    for (int i = 0; i < vper; ++i) {
        int pk = 0;
#pragma unroll
        for (int j = 0; j < 4; ++j) {
            float qv = rintf(v[4*i+j] * scale);
            qv = fminf(127.f, fmaxf(-128.f, qv));
            pk |= ((int)qv & 255) << (8 * j);
        }
        ((int*)Q)[row * (long)(W >> 2) + i * 256 + t] = pk;
    }
    if (t == 0) inv_scale[row] = mx / 127.0f;
}

template <int NTILES, bool HAS_RESID>
__global__ void gemm_i8_wmma(const int8_t* __restrict__ A,
                             const int8_t* __restrict__ Bw,
                             const float* __restrict__ invs,
                             const float* __restrict__ resid,
                             float* __restrict__ C,
                             int M, int N, int K) {
    const int lane = threadIdx.x & 31;
    const int wave = threadIdx.x >> 5;
    const int ngroups = N / (16 * NTILES);
    const int total = (M >> 4) * ngroups;
    const int wg = blockIdx.x * 8 + wave;
    if (wg >= total) return;

    const int m0 = (wg / ngroups) << 4;
    const int n0 = (wg % ngroups) * (16 * NTILES);
    const int r  = lane & 15;
    const int hi = lane >> 4;

    const int8_t* arow  = A  + (long)(m0 + r) * K;
    const int8_t* brow0 = Bw + (long)(n0 + r) * K;

    v8i acc[NTILES];
#pragma unroll
    for (int j = 0; j < NTILES; ++j) acc[j] = (v8i){0,0,0,0,0,0,0,0};

    for (int k0 = 0; k0 < K; k0 += 64) {
        v8i a;
#pragma unroll
        for (int v = 0; v < 8; ++v) {
            const int ka = k0 + ((v >> 1) << 4) + ((v & 1) << 2) + (hi << 3);
            a[v] = *(const int*)(arow + ka);
        }
        __builtin_prefetch(arow + k0 + 64, 0, 1);
#pragma unroll
        for (int j = 0; j < NTILES; ++j) {
            const int8_t* brow = brow0 + (long)(j * 16) * K;
            v8i b;
#pragma unroll
            for (int v = 0; v < 8; ++v) {
                const int kb = k0 + ((v >> 1) << 4) + ((v & 1) << 2) + (hi << 3);
                b[v] = *(const int*)(brow + kb);
            }
            acc[j] = __builtin_amdgcn_wmma_i32_16x16x64_iu8(
                 true, a,  true, b, acc[j],
                 false,  false);
        }
    }

#pragma unroll
    for (int j = 0; j < NTILES; ++j) {
        const int n = n0 + j * 16 + r;
#pragma unroll
        for (int q = 0; q < 8; ++q) {
            const int m = m0 + q + (hi << 3);
            float val = (float)acc[j][q] * invs[m];
            const long idx = (long)m * N + n;
            if (HAS_RESID) val += resid[idx];
            C[idx] = val;
        }
    }
}

__global__ void conv_silu(const float* __restrict__ xz,
                          const float* __restrict__ cw,
                          const float* __restrict__ cb,
                          float* __restrict__ xc) {
    const long gid = (long)blockIdx.x * 256 + threadIdx.x;
    if (gid >= (long)ROWS * D_INNER) return;
    const int d  = (int)(gid % D_INNER);
    const long rr = gid / D_INNER;
    const int l  = (int)(rr % SEQ);
    const int b  = (int)(rr / SEQ);
    float acc = cb[d];
#pragma unroll
    for (int j = 0; j < D_CONV; ++j) {
        const int li = l - (D_CONV - 1) + j;
        if (li >= 0)
            acc += cw[d * D_CONV + j] *
                   xz[((long)b * SEQ + li) * (2 * D_INNER) + d];
    }
    xc[gid] = acc / (1.0f + expf(-acc));
}

__global__ void dt_delta(const float* __restrict__ xdbl,
                         const float* __restrict__ dtw,
                         const float* __restrict__ dtb,
                         float* __restrict__ delta) {
    const long gid = (long)blockIdx.x * 256 + threadIdx.x;
    if (gid >= (long)ROWS * D_INNER) return;
    const int d  = (int)(gid % D_INNER);
    const long rr = gid / D_INNER;
    const float4* x4 = (const float4*)(xdbl + rr * XDBL_W);
    const float4* w4 = (const float4*)(dtw + (long)d * DT_RANK);
    float s = dtb[d];
#pragma unroll
    for (int k = 0; k < DT_RANK / 4; ++k) {
        float4 a = x4[k], b = w4[k];
        s += a.x * b.x + a.y * b.y + a.z * b.z + a.w * b.w;
    }
    const float v = s + dtb[d];
    delta[gid] = (v > 20.f) ? v : log1pf(expf(v));
}

__global__ void scan_kernel(const float* __restrict__ xc,
                            const float* __restrict__ delta,
                            const float* __restrict__ xdbl,
                            const float* __restrict__ A_log,
                            const float* __restrict__ Dp,
                            const float* __restrict__ xz,
                            float* __restrict__ y) {
    const int gid = blockIdx.x * 256 + threadIdx.x;
    if (gid >= BATCH * D_INNER * D_STATE) return;
    const int n  = gid & (D_STATE - 1);
    const int bd = gid >> 4;
    const int d  = bd & (D_INNER - 1);
    const int b  = bd >> 11;

    const float Ad = -expf(A_log[(long)d * D_STATE + n]);
    const float Dd = Dp[d];
    float h = 0.f;

    for (int l = 0; l < SEQ; ++l) {
        const long row = (long)b * SEQ + l;
        const float dt = delta[row * D_INNER + d];
        const float xv = xc[row * D_INNER + d];
        const float Bn = xdbl[row * XDBL_W + DT_RANK + n];
        const float Cn = xdbl[row * XDBL_W + DT_RANK + D_STATE + n];
        h = expf(dt * Ad) * h + dt * Bn * xv;
        float p = h * Cn;
        p += __shfl_xor(p, 1, 16);
        p += __shfl_xor(p, 2, 16);
        p += __shfl_xor(p, 4, 16);
        p += __shfl_xor(p, 8, 16);
        if (n == 0) {
            const float zv = xz[row * (2 * D_INNER) + D_INNER + d];
            y[row * D_INNER + d] = (p + xv * Dd) * (zv / (1.0f + expf(-zv)));
        }
    }
}

extern "C" void kernel_launch(void* const* d_in, const int* in_sizes, int n_in,
                              void* d_out, int out_size, void* d_ws, size_t ws_size,
                              hipStream_t stream) {
    (void)in_sizes; (void)n_in; (void)out_size; (void)ws_size;

    const float* hidden  = (const float*)d_in[0];
    const float* rms_w   = (const float*)d_in[1];
    const float* in_w    = (const float*)d_in[2];
    const float* xp_w    = (const float*)d_in[3];
    const float* out_w   = (const float*)d_in[4];
    const float* conv_w  = (const float*)d_in[5];
    const float* conv_b  = (const float*)d_in[6];
    const float* dt_w    = (const float*)d_in[7];
    const float* dt_b    = (const float*)d_in[8];
    const float* A_log   = (const float*)d_in[9];
    const float* Dp      = (const float*)d_in[10];
    float* out           = (float*)d_out;

    char* base = (char*)d_ws;
    size_t off = 0;
    auto alloc = [&](size_t bytes) -> void* {
        void* p = (void*)(base + off);
        off += (bytes + 255) & ~(size_t)255;
        return p;
    };
    float*  partial = (float*)alloc(3 * 256 * sizeof(float));
    float*  wscale  = (float*)alloc(3 * sizeof(float));
    float*  inv1    = (float*)alloc(ROWS * sizeof(float));
    float*  inv2    = (float*)alloc(ROWS * sizeof(float));
    float*  inv3    = (float*)alloc(ROWS * sizeof(float));
    int8_t* A8      = (int8_t*)alloc((size_t)ROWS * D_INNER);
    int8_t* W8_in   = (int8_t*)alloc((size_t)(2 * D_INNER) * DIM);
    int8_t* W8_xp   = (int8_t*)alloc((size_t)XDBL_W * D_INNER);
    int8_t* W8_out  = (int8_t*)alloc((size_t)DIM * D_INNER);
    float*  xz      = (float*)alloc((size_t)ROWS * 2 * D_INNER * sizeof(float));
    float*  xconv   = (float*)alloc((size_t)ROWS * D_INNER * sizeof(float));
    float*  xdbl    = (float*)alloc((size_t)ROWS * XDBL_W * sizeof(float));
    float*  delta   = (float*)alloc((size_t)ROWS * D_INNER * sizeof(float));
    float*  ybuf    = (float*)alloc((size_t)ROWS * D_INNER * sizeof(float));

    const long n_in_w  = (long)(2 * D_INNER) * DIM;
    const long n_xp_w  = (long)XDBL_W * D_INNER;
    const long n_out_w = (long)DIM * D_INNER;

    absmean_partial<<<256, 256, 0, stream>>>(in_w,  n_in_w,  partial + 0);
    absmean_partial<<<256, 256, 0, stream>>>(xp_w,  n_xp_w,  partial + 256);
    absmean_partial<<<256, 256, 0, stream>>>(out_w, n_out_w, partial + 512);
    finalize_scales<<<1, 256, 0, stream>>>(partial, wscale);
    quant_weight<<<2048, 256, 0, stream>>>(in_w,  n_in_w,  wscale + 0, W8_in);
    quant_weight<<<2048, 256, 0, stream>>>(xp_w,  n_xp_w,  wscale + 1, W8_xp);
    quant_weight<<<2048, 256, 0, stream>>>(out_w, n_out_w, wscale + 2, W8_out);

    ln_act_quant<<<ROWS, 256, 0, stream>>>(hidden, DIM, rms_w, 1, A8, inv1);

    {
        const int total = (ROWS / 16) * ((2 * D_INNER) / 64);
        gemm_i8_wmma<4, false><<<(total + 7) / 8, 256, 0, stream>>>(
            A8, W8_in, inv1, nullptr, xz, ROWS, 2 * D_INNER, DIM);
    }

    conv_silu<<<(int)(((long)ROWS * D_INNER + 255) / 256), 256, 0, stream>>>(
        xz, conv_w, conv_b, xconv);

    ln_act_quant<<<ROWS, 256, 0, stream>>>(xconv, D_INNER, nullptr, 0, A8, inv2);

    {
        const int total = (ROWS / 16) * (XDBL_W / 32);
        gemm_i8_wmma<2, false><<<(total + 7) / 8, 256, 0, stream>>>(
            A8, W8_xp, inv2, nullptr, xdbl, ROWS, XDBL_W, D_INNER);
    }

    dt_delta<<<(int)(((long)ROWS * D_INNER + 255) / 256), 256, 0, stream>>>(
        xdbl, dt_w, dt_b, delta);

    scan_kernel<<<(BATCH * D_INNER * D_STATE + 255) / 256, 256, 0, stream>>>(
        xconv, delta, xdbl, A_log, Dp, xz, ybuf);

    ln_act_quant<<<ROWS, 256, 0, stream>>>(ybuf, D_INNER, nullptr, 0, A8, inv3);

    {
        const int total = (ROWS / 16) * (DIM / 64);
        gemm_i8_wmma<4, true><<<(total + 7) / 8, 256, 0, stream>>>(
            A8, W8_out, inv3, hidden, out, ROWS, DIM, D_INNER);
    }
}
